// RGCN_45260365365441
// MI455X (gfx1250) — matrix operand order repaired and checked by reading, not compiled, not run on hardware
//
#include <hip/hip_runtime.h>

// MI455X / gfx1250, wave32. WMMA f16->f32 GEMMs, bucketed per-relation scatter,
// f16 message buffers (L2-resident), degree-folded mean aggregation.

typedef _Float16 half8 __attribute__((ext_vector_type(8)));
typedef _Float16 v16h  __attribute__((ext_vector_type(16)));
typedef float    v8f   __attribute__((ext_vector_type(8)));

#define N_NODES 100000
#define N_EDGES 1000000
#define N_REL   8
#define N_BASES 30
#define D_IN    128
#define NUM_OUT 50000

// ---------------- utility kernels ----------------

__global__ void k_zero_f32(float* __restrict__ p, size_t n) {
  size_t i  = (size_t)blockIdx.x * blockDim.x + threadIdx.x;
  size_t st = (size_t)gridDim.x * blockDim.x;
  for (; i < n; i += st) p[i] = 0.0f;
}

__global__ void k_zero_i32(int* __restrict__ p, int n) {
  int i = blockIdx.x * blockDim.x + threadIdx.x;
  if (i < n) p[i] = 0;
}

__global__ void k_f32_to_f16(const float* __restrict__ in, _Float16* __restrict__ out, size_t n) {
  size_t i  = (size_t)blockIdx.x * blockDim.x + threadIdx.x;
  size_t st = (size_t)gridDim.x * blockDim.x;
  for (; i < n; i += st) out[i] = (_Float16)in[i];
}

__global__ void k_copy_f32(const float* __restrict__ in, float* __restrict__ out, size_t n) {
  size_t i  = (size_t)blockIdx.x * blockDim.x + threadIdx.x;
  size_t st = (size_t)gridDim.x * blockDim.x;
  for (; i < n; i += st) out[i] = in[i];
}

// wt[r][f][k] = sum_b comp[r,b] * basis[b][k][f]   (f16, pre-transposed for B-fragment loads)
__global__ void k_make_wt(const float* __restrict__ comp, const float* __restrict__ basis,
                          _Float16* __restrict__ wt, int fout) {
  int idx = blockIdx.x * blockDim.x + threadIdx.x;
  int tot = N_REL * fout * D_IN;
  if (idx >= tot) return;
  int k = idx % D_IN;
  int f = (idx / D_IN) % fout;
  int r = idx / (D_IN * fout);
  float acc = 0.0f;
  for (int b = 0; b < N_BASES; ++b)
    acc += comp[r * N_BASES + b] * basis[((size_t)b * D_IN + k) * fout + f];
  wt[idx] = (_Float16)acc;
}

// dst[f][k] = src[k][f]  (f32 [K,F] -> f16 [F,K])
__global__ void k_transpose_h(const float* __restrict__ src, _Float16* __restrict__ dst,
                              int K, int F) {
  int idx = blockIdx.x * blockDim.x + threadIdx.x;
  if (idx >= K * F) return;
  int k = idx % K;
  int f = idx / K;
  dst[idx] = (_Float16)src[k * F + f];
}

// ---------------- degree + edge bucketing (counting sort by relation) ----------------

__global__ void k_deg(const int* __restrict__ ei, const int* __restrict__ et,
                      float* __restrict__ deg) {
  size_t e  = (size_t)blockIdx.x * blockDim.x + threadIdx.x;
  size_t st = (size_t)gridDim.x * blockDim.x;
  for (; e < N_EDGES; e += st)
    atomicAdd(&deg[(size_t)et[e] * N_NODES + ei[N_EDGES + e]], 1.0f);
}

__global__ void k_count(const int* __restrict__ et, int* __restrict__ cnt) {
  size_t e  = (size_t)blockIdx.x * blockDim.x + threadIdx.x;
  size_t st = (size_t)gridDim.x * blockDim.x;
  for (; e < N_EDGES; e += st) atomicAdd(&cnt[et[e]], 1);
}

__global__ void k_prefix(const int* __restrict__ cnt, int* __restrict__ offs,
                         int* __restrict__ cur) {
  if (threadIdx.x == 0 && blockIdx.x == 0) {
    int acc = 0;
    for (int r = 0; r < N_REL; ++r) {
      offs[r] = acc;
      cur[r]  = acc;
      acc += cnt[r];
    }
    offs[N_REL] = acc;
  }
}

__global__ void k_fill(const int* __restrict__ et, int* __restrict__ cur,
                       int* __restrict__ bucket) {
  size_t e  = (size_t)blockIdx.x * blockDim.x + threadIdx.x;
  size_t st = (size_t)gridDim.x * blockDim.x;
  for (; e < N_EDGES; e += st) {
    int r = et[e];
    int pos = atomicAdd(&cur[r], 1);
    bucket[pos] = (int)e;
  }
}

// ---------------- WMMA GEMM: C[rows,F] = A[rows,128] * Wt^T (+bias) ----------------
// A: f16 row-major [rows,128]; Wt: f16 [F,128] (B transposed); C: f32 or f16.
// One wave owns a 16-row strip and all F columns (NT = F/16 accumulator tiles).
// Weight matrix staged in LDS once per block. In-place A==C is safe: each row is
// read only by the wave that writes it; clamped padding reads feed only rows that
// are never stored.
template <int NT, typename OutT>
__global__ void __launch_bounds__(128)
k_wmma_gemm(const _Float16* A, const _Float16* __restrict__ Wt,
            const float* __restrict__ bias, OutT* C, int rows) {
  constexpr int K = 128;
  constexpr int F = NT * 16;
  __shared__ alignas(32) _Float16 sW[F * K];

  {
    const half8* src = (const half8*)Wt;
    half8*       dst = (half8*)sW;
    constexpr int chunks = F * K / 8;
    for (int i = threadIdx.x; i < chunks; i += 128) dst[i] = src[i];
  }
  __syncthreads();

  const int lane = threadIdx.x & 31;
  const int wave = threadIdx.x >> 5;
  const int m0   = (blockIdx.x * 4 + wave) * 16;
  if (m0 < rows) {                      // wave-uniform: EXEC stays all-ones for WMMA
    const int lr = lane & 15;           // row-in-tile (A) / col-in-tile (B,C)
    const int hi = lane >> 4;           // lane-half select

    int arow = m0 + lr;
    if (arow > rows - 1) arow = rows - 1;
    const _Float16* aptr = A + (size_t)arow * K;

    v8f acc[NT];
#pragma unroll
    for (int t = 0; t < NT; ++t) {
      float bv = bias ? bias[t * 16 + lr] : 0.0f;
      v8f v = {bv, bv, bv, bv, bv, bv, bv, bv};
      acc[t] = v;
    }

#pragma unroll
    for (int k0 = 0; k0 < K; k0 += 32) {
      // A fragment (ISA 16-bit A 16x32): halves 0-7 = K kb..kb+7, 8-15 = kb+16..kb+23
      const int kb = k0 + hi * 8;
      half8 alo = *(const half8*)(aptr + kb);
      half8 ahi = *(const half8*)(aptr + kb + 16);
      v16h a = __builtin_shufflevector(alo, ahi, 0, 1, 2, 3, 4, 5, 6, 7,
                                       8, 9, 10, 11, 12, 13, 14, 15);
#pragma unroll
      for (int t = 0; t < NT; ++t) {
        const half8 blo = *(const half8*)(sW + ((t * 16 + lr) * K + kb));
        const half8 bhi = *(const half8*)(sW + ((t * 16 + lr) * K + kb + 16));
        const v16h b = __builtin_shufflevector(blo, bhi, 0, 1, 2, 3, 4, 5, 6, 7,
                                               8, 9, 10, 11, 12, 13, 14, 15);
        acc[t] = __builtin_amdgcn_wmma_f32_16x16x32_f16(
            false, a, false, b, (short)0, acc[t], false, false);
      }
    }

    // C layout: VGPR v, lane l -> row m0 + v + 8*lanehalf, col = t*16 + lr
#pragma unroll
    for (int t = 0; t < NT; ++t) {
#pragma unroll
      for (int v = 0; v < 8; ++v) {
        int row = m0 + v + hi * 8;
        if (row < rows) C[(size_t)row * F + t * 16 + lr] = (OutT)acc[t][v];
      }
    }
  }
}

// ---------------- bucketed edge scatter: out[dst] += h[src] / deg[r][dst] ----------------
// Fixed-size grid-stride over the relation's compacted edge list; edge count read
// on-device (graph-capture safe, no host readback).
template <int FDIM>
__global__ void __launch_bounds__(256)
k_scatter(const int* __restrict__ ei, const int* __restrict__ bucket,
          const int* __restrict__ offs, const _Float16* __restrict__ h,
          const float* __restrict__ deg, float* __restrict__ out, int r) {
  const int base = offs[r];
  const int n    = offs[r + 1] - base;
  constexpr int SLOTS = 256 / FDIM;           // edges handled per block iteration
  const int f    = threadIdx.x % FDIM;
  const int slot = threadIdx.x / FDIM;
  int i = blockIdx.x * SLOTS + slot;
  const int stride = gridDim.x * SLOTS;
  for (; i < n; i += stride) {
    int e   = bucket[base + i];
    int src = ei[e];
    int dst = ei[N_EDGES + e];
    float s = 1.0f / deg[(size_t)r * N_NODES + dst];   // deg >= 1 for any matched edge
    atomicAdd(&out[(size_t)dst * FDIM + f],
              (float)h[(size_t)src * FDIM + f] * s);
  }
}

// ---------------- driver ----------------

extern "C" void kernel_launch(void* const* d_in, const int* in_sizes, int n_in,
                              void* d_out, int out_size, void* d_ws, size_t ws_size,
                              hipStream_t stream) {
  const float* x_raw  = (const float*)d_in[0];
  const int*   ei     = (const int*)d_in[1];   // [2,E]
  const int*   et     = (const int*)d_in[2];   // [E]
  const float* lin_w  = (const float*)d_in[3]; // [128,128] already [F][K]
  const float* lin_b  = (const float*)d_in[4];
  const float* basis1 = (const float*)d_in[5];
  const float* comp1  = (const float*)d_in[6];
  const float* root1  = (const float*)d_in[7];
  const float* bias1  = (const float*)d_in[8];
  const float* basis2 = (const float*)d_in[9];
  const float* comp2  = (const float*)d_in[10];
  const float* root2  = (const float*)d_in[11];
  const float* bias2  = (const float*)d_in[12];

  char*  ws  = (char*)d_ws;
  size_t off = 0;
  auto carve = [&](size_t bytes) -> void* {
    off = (off + 255) & ~(size_t)255;
    void* p = ws + off;
    off += bytes;
    return p;
  };

  _Float16* xh     = (_Float16*)carve((size_t)N_NODES * 128 * 2); // f16 layer input
  float*    bufA   = (float*)carve((size_t)N_NODES * 128 * 4);    // f32 layer output acc
  _Float16* hh     = (_Float16*)carve((size_t)N_NODES * 128 * 2); // f16 per-relation messages
  float*    deg    = (float*)carve((size_t)N_REL * N_NODES * 4);
  int*      bucket = (int*)carve((size_t)N_EDGES * 4);
  int*      cnt    = (int*)carve(64);
  int*      offs   = (int*)carve(64);
  int*      cur    = (int*)carve(64);
  _Float16* lin_t  = (_Float16*)carve((size_t)128 * 128 * 2);
  _Float16* w1t    = (_Float16*)carve((size_t)N_REL * 128 * 128 * 2);
  _Float16* r1t    = (_Float16*)carve((size_t)128 * 128 * 2);
  _Float16* w2t    = (_Float16*)carve((size_t)N_REL * 64 * 128 * 2);
  _Float16* r2t    = (_Float16*)carve((size_t)64 * 128 * 2);

  // per-(relation,dst) degree table + counting sort of edges by relation
  k_zero_f32<<<1024, 256, 0, stream>>>(deg, (size_t)N_REL * N_NODES);
  k_zero_i32<<<1, 32, 0, stream>>>(cnt, N_REL);
  k_deg<<<2048, 256, 0, stream>>>(ei, et, deg);
  k_count<<<2048, 256, 0, stream>>>(et, cnt);
  k_prefix<<<1, 1, 0, stream>>>(cnt, offs, cur);
  k_fill<<<2048, 256, 0, stream>>>(et, cur, bucket);

  // weight preparation (tiny)
  k_f32_to_f16<<<64, 256, 0, stream>>>(lin_w, lin_t, (size_t)128 * 128);
  k_make_wt<<<(N_REL * 128 * 128 + 255) / 256, 256, 0, stream>>>(comp1, basis1, w1t, 128);
  k_transpose_h<<<(128 * 128 + 255) / 256, 256, 0, stream>>>(root1, r1t, 128, 128);
  k_make_wt<<<(N_REL * 64 * 128 + 255) / 256, 256, 0, stream>>>(comp2, basis2, w2t, 64);
  k_transpose_h<<<(128 * 64 + 255) / 256, 256, 0, stream>>>(root2, r2t, 128, 64);

  const int gemmGrid = (N_NODES + 63) / 64;   // 4 waves/block, 16 rows/wave

  // stage 0: x = x_raw @ lin_w^T + lin_b   (f16 output written in-place into xh)
  k_f32_to_f16<<<4096, 256, 0, stream>>>(x_raw, xh, (size_t)N_NODES * 128);
  k_wmma_gemm<8, _Float16><<<gemmGrid, 128, 0, stream>>>(xh, lin_t, lin_b, xh, N_NODES);

  // layer 1 (F=128)
  k_wmma_gemm<8, float><<<gemmGrid, 128, 0, stream>>>(xh, r1t, bias1, bufA, N_NODES);
  for (int r = 0; r < N_REL; ++r) {
    k_wmma_gemm<8, _Float16><<<gemmGrid, 128, 0, stream>>>(
        xh, w1t + (size_t)r * 128 * 128, nullptr, hh, N_NODES);
    k_scatter<128><<<2048, 256, 0, stream>>>(ei, bucket, offs, hh, deg, bufA, r);
  }
  k_f32_to_f16<<<4096, 256, 0, stream>>>(bufA, xh, (size_t)N_NODES * 128);

  // layer 2 (F=64)
  k_wmma_gemm<4, float><<<gemmGrid, 128, 0, stream>>>(xh, r2t, bias2, bufA, N_NODES);
  for (int r = 0; r < N_REL; ++r) {
    k_wmma_gemm<4, _Float16><<<gemmGrid, 128, 0, stream>>>(
        xh, w2t + (size_t)r * 64 * 128, nullptr, hh, N_NODES);
    k_scatter<64><<<2048, 256, 0, stream>>>(ei, bucket, offs, hh, deg, bufA, r);
  }

  // out = x[:NUM_OUT]
  k_copy_f32<<<4096, 256, 0, stream>>>(bufA, (float*)d_out, (size_t)NUM_OUT * 64);
}
